// crossatt_76544907149890
// MI455X (gfx1250) — matrix operand order repaired and checked by reading, not compiled, not run on hardware
//
#include <hip/hip_runtime.h>
#include <hip/hip_bf16.h>

// ---------- types ----------
typedef __attribute__((ext_vector_type(16))) __bf16        v16bf;
typedef __attribute__((ext_vector_type(4)))  __bf16        v4bf;
typedef __attribute__((ext_vector_type(8)))  float         v8f;
typedef __attribute__((ext_vector_type(4)))  float         v4f;
typedef __attribute__((ext_vector_type(4)))  unsigned int  v4u;

union Frag16 { v16bf v; v4u u[2]; };

#define B_  32
#define LD_ 2048
#define LQ_ 512
#define D_  256

// native TANH trans op on gfx1250 if the builtin exists, else OCML fallback
#if __has_builtin(__builtin_amdgcn_tanhf)
#define TANH_F32(x) __builtin_amdgcn_tanhf(x)
#else
#define TANH_F32(x) tanhf(x)
#endif

// ---------------------------------------------------------------
// Kernel 0: zero the reduction workspace
// ---------------------------------------------------------------
__global__ void zero_ws_kernel(float* __restrict__ p, int n) {
    int i = blockIdx.x * 256 + threadIdx.x;
    if (i < n) p[i] = 0.0f;
}

// ---------------------------------------------------------------
// Kernel 1: S = tanh(Q . Doc^T) with fused row/col reductions.
//   grid = (Ld/128, Lq/128, B), block = 256 (8 waves, 4(M) x 2(N))
//   rowsum[b][t] += sum_q tanh(s)   (doc-axis scores)
//   colsum[b][q] += sum_t tanh(s)   (query-axis scores)
// ---------------------------------------------------------------
__global__ void __launch_bounds__(256)
gemm_tanh_reduce_kernel(const float* __restrict__ doc,
                        const float* __restrict__ query,
                        float* __restrict__ rowsum,   // [B, LD_]
                        float* __restrict__ colsum) { // [B, LQ_]
    // padded LDS: row stride 40 bf16 (80B) -> conflict-free b128 frag loads
    __shared__ __align__(16) __bf16 Qs[128 * 40];
    __shared__ __align__(16) __bf16 Ds[128 * 40];

    const int tid   = threadIdx.x;
    const int lane  = tid & 31;
    const int wave  = tid >> 5;
    const int b     = blockIdx.z;
    const int qBase = blockIdx.y * 128;   // query rows (M)
    const int tBase = blockIdx.x * 128;   // doc rows   (N)

    const int mBase = (wave & 3) * 32;    // 4 waves along M: 32 rows each
    const int nBase = (wave >> 2) * 64;   // 2 waves along N: 64 cols each

    const float* Qg = query + (size_t)b * LQ_ * D_;
    const float* Dg = doc   + (size_t)b * LD_ * D_;

    const int m  = lane & 15;
    const int kh = lane >> 4;

    v8f acc[2][4];
#pragma unroll
    for (int i = 0; i < 2; ++i)
#pragma unroll
        for (int j = 0; j < 4; ++j) acc[i][j] = (v8f){0,0,0,0,0,0,0,0};

    for (int kk = 0; kk < D_; kk += 32) {
        __syncthreads();
        // ---- stage 128x32 f32 tiles of Q and Doc into LDS as bf16 ----
        // hardware f32->bf16 converts via fptrunc (no manual bit-twiddling)
#pragma unroll
        for (int i = 0; i < 4; ++i) {
            int idx = tid + i * 256;       // 0..1023
            int row = idx >> 3;            // 0..127
            int c4  = (idx & 7) << 2;      // 0,4,...,28

            v4f qv = *(const v4f*)(Qg + (size_t)(qBase + row) * D_ + kk + c4);
            v4bf qb = { (__bf16)qv.x, (__bf16)qv.y, (__bf16)qv.z, (__bf16)qv.w };
            *(v4bf*)&Qs[row * 40 + c4] = qb;

            v4f dv = *(const v4f*)(Dg + (size_t)(tBase + row) * D_ + kk + c4);
            v4bf db = { (__bf16)dv.x, (__bf16)dv.y, (__bf16)dv.z, (__bf16)dv.w };
            *(v4bf*)&Ds[row * 40 + c4] = db;
        }
        __syncthreads();

        // ---- build fragments (ISA 7.12.2 layouts) ----
        Frag16 fa[2], fb[4];
#pragma unroll
        for (int i = 0; i < 2; ++i) {   // A 16x32: two 8-elem K chunks / lane
            const __bf16* p = &Qs[(mBase + i * 16 + m) * 40];
            fa[i].u[0] = *(const v4u*)(p + kh * 8);
            fa[i].u[1] = *(const v4u*)(p + 16 + kh * 8);
        }
#pragma unroll
        for (int j = 0; j < 4; ++j) {   // B 32x16: two 8-elem K chunks / lane
            const __bf16* p = &Ds[(nBase + j * 16 + m) * 40];
            fb[j].u[0] = *(const v4u*)(p + kh * 8);
            fb[j].u[1] = *(const v4u*)(p + 16 + kh * 8);
        }
#pragma unroll
        for (int i = 0; i < 2; ++i)
#pragma unroll
            for (int j = 0; j < 4; ++j)
                acc[i][j] = __builtin_amdgcn_wmma_f32_16x16x32_bf16(
                    false, fa[i].v, false, fb[j].v,
                    (short)0, acc[i][j], false, false);
    }

    // ---- tanh + fused reductions ----
    // C layout: reg r, lane L -> (M = r + 8*(L>>4), N = L&15)
    float rowAcc[4] = {0.f, 0.f, 0.f, 0.f};   // per n-subtile, per lane (fixed N)
#pragma unroll
    for (int i = 0; i < 2; ++i) {
        float colAcc[8] = {0.f,0.f,0.f,0.f,0.f,0.f,0.f,0.f};
#pragma unroll
        for (int j = 0; j < 4; ++j) {
            float lanesum = 0.f;
#pragma unroll
            for (int r = 0; r < 8; ++r) {
                float tv = TANH_F32(acc[i][j][r]);
                lanesum   += tv;   // partial sum over this lane's 8 M values
                colAcc[r] += tv;   // partial sum over N (this lane's N, all j)
            }
            rowAcc[j] += lanesum;
        }
        // reduce colAcc over the 16 lanes of each half (all share the M set)
#pragma unroll
        for (int r = 0; r < 8; ++r) {
            float v = colAcc[r];
            v += __shfl_xor(v, 1);
            v += __shfl_xor(v, 2);
            v += __shfl_xor(v, 4);
            v += __shfl_xor(v, 8);
            if ((lane & 15) == 0) {
                int q = qBase + mBase + i * 16 + r + 8 * kh;
                atomicAdd(&colsum[(size_t)b * LQ_ + q], v);
            }
        }
    }
    // combine the two M-halves (lane ^ 16 has same N) then one atomic per N
#pragma unroll
    for (int j = 0; j < 4; ++j) {
        float v = rowAcc[j] + __shfl_xor(rowAcc[j], 16);
        if (lane < 16) {
            int t = tBase + nBase + j * 16 + lane;
            atomicAdd(&rowsum[(size_t)b * LD_ + t], v);
        }
    }
}

// ---------------------------------------------------------------
// Kernel 2: in-place masked softmax over the last axis.
//   buf[b, :] = exp(x - max) * mask / sum(exp(x - max) * mask)
//   (mask-then-renormalize == softmax * mask renormalized)
//   grid = B, block = 256, L in {2048, 512}
// ---------------------------------------------------------------
__global__ void __launch_bounds__(256)
softmax_mask_kernel(float* __restrict__ buf, const float* __restrict__ mask, int L) {
    __shared__ float red[256];
    const int b   = blockIdx.x;
    const int tid = threadIdx.x;
    float* row        = buf  + (size_t)b * L;
    const float* mrow = mask + (size_t)b * L;

    const int n = L >> 8;            // 8 or 2 items / thread
    float v[8];
    float mx = -3.402823466e38f;
    for (int i = 0; i < n; ++i) { v[i] = row[tid + i * 256]; mx = fmaxf(mx, v[i]); }

    red[tid] = mx; __syncthreads();
    for (int s = 128; s > 0; s >>= 1) {
        if (tid < s) red[tid] = fmaxf(red[tid], red[tid + s]);
        __syncthreads();
    }
    mx = red[0]; __syncthreads();

    float sum = 0.f;
    for (int i = 0; i < n; ++i) {
        float e = __expf(v[i] - mx) * mrow[tid + i * 256];
        v[i] = e; sum += e;
    }
    red[tid] = sum; __syncthreads();
    for (int s = 128; s > 0; s >>= 1) {
        if (tid < s) red[tid] += red[tid + s];
        __syncthreads();
    }
    float inv = 1.0f / red[0];
    for (int i = 0; i < n; ++i) row[tid + i * 256] = v[i] * inv;
}

// ---------------------------------------------------------------
// Kernel 3: weighted pooling + concat.
//   out[b, 0:256]   = sum_t alpha[b,t] * doc[b,t,:]
//   out[b, 256:512] = sum_q beta [b,q] * query[b,q,:]
//   grid = B, block = 256 (one thread per feature d)
// ---------------------------------------------------------------
__global__ void __launch_bounds__(256)
pool_concat_kernel(const float* __restrict__ doc, const float* __restrict__ query,
                   const float* __restrict__ alpha, const float* __restrict__ beta,
                   float* __restrict__ out) {
    const int b = blockIdx.x;
    const int d = threadIdx.x;
    const float* db = doc   + (size_t)b * LD_ * D_;
    const float* qb = query + (size_t)b * LQ_ * D_;
    const float* al = alpha + (size_t)b * LD_;
    const float* be = beta  + (size_t)b * LQ_;

    float accD = 0.f;
    for (int t = 0; t < LD_; t += 4) {
        accD += al[t    ] * db[(size_t)(t    ) * D_ + d];
        accD += al[t + 1] * db[(size_t)(t + 1) * D_ + d];
        accD += al[t + 2] * db[(size_t)(t + 2) * D_ + d];
        accD += al[t + 3] * db[(size_t)(t + 3) * D_ + d];
    }
    float accQ = 0.f;
    for (int q = 0; q < LQ_; q += 4) {
        accQ += be[q    ] * qb[(size_t)(q    ) * D_ + d];
        accQ += be[q + 1] * qb[(size_t)(q + 1) * D_ + d];
        accQ += be[q + 2] * qb[(size_t)(q + 2) * D_ + d];
        accQ += be[q + 3] * qb[(size_t)(q + 3) * D_ + d];
    }
    out[(size_t)b * 512 + d]       = accD;
    out[(size_t)b * 512 + 256 + d] = accQ;
}

// ---------------------------------------------------------------
extern "C" void kernel_launch(void* const* d_in, const int* in_sizes, int n_in,
                              void* d_out, int out_size, void* d_ws, size_t ws_size,
                              hipStream_t stream) {
    const float* doc      = (const float*)d_in[0];  // [32,2048,256]
    const float* query    = (const float*)d_in[1];  // [32, 512,256]
    const float* doc_mask = (const float*)d_in[2];  // [32,2048]
    const float* q_mask   = (const float*)d_in[3];  // [32, 512]
    float* out = (float*)d_out;                     // [32, 512]

    float* rowsum = (float*)d_ws;          // [32,2048] -> becomes alpha
    float* colsum = rowsum + B_ * LD_;     // [32, 512] -> becomes beta

    const int nred = B_ * LD_ + B_ * LQ_;  // 81920 floats (320 KB)
    zero_ws_kernel<<<(nred + 255) / 256, 256, 0, stream>>>(rowsum, nred);

    dim3 grid(LD_ / 128, LQ_ / 128, B_);   // (16, 4, 32) = 2048 workgroups
    gemm_tanh_reduce_kernel<<<grid, 256, 0, stream>>>(doc, query, rowsum, colsum);

    softmax_mask_kernel<<<B_, 256, 0, stream>>>(rowsum, doc_mask, LD_);
    softmax_mask_kernel<<<B_, 256, 0, stream>>>(colsum, q_mask,  LQ_);

    pool_concat_kernel<<<B_, 256, 0, stream>>>(doc, query, rowsum, colsum, out);
}
